// SimpleRNN_22660247454198
// MI455X (gfx1250) — matrix operand order repaired and checked by reading, not compiled, not run on hardware
//
#include <hip/hip_runtime.h>

typedef __attribute__((ext_vector_type(16))) _Float16 v16h;
typedef __attribute__((ext_vector_type(8)))  float    v8f;

#define B_TOT   4096
#define K_STEPS 512
#define U_IN    3
#define P_ST    5
#define LIFT    32
#define HID     128
#define G3      (3 * HID)   // 384
#define NSUB    10
#define TM      64          // batch rows per workgroup
#define THREADS 256         // 8 waves (wave32)
#define NBLK    (B_TOT / TM)

// ---------------- LDS layout (byte offsets, all 64-aligned) ----------------
#define OFF_WHH   0                      // 384*128 f16 = 98304
#define OFF_WIH   (OFF_WHH + 98304)     // 384*32  f16 = 24576
#define OFF_WHD   (OFF_WIH + 24576)     // 16*128  f16 = 4096 (Wh padded 13->16)
#define OFF_X     (OFF_WHD + 4096)      // 64*32   f16 = 4096
#define OFF_H0    (OFF_X   + 4096)      // 64*128  f16 = 16384
#define OFF_H1    (OFF_H0  + 16384)     // 64*128  f16 = 16384
#define OFF_WL    (OFF_H1  + 16384)     // 32*8    f32 = 1024
#define OFF_BIH   (OFF_WL  + 1024)      // 384     f32 = 1536
#define OFF_BHH   (OFF_BIH + 1536)      // 384     f32 = 1536
#define OFF_BL    (OFF_BHH + 1536)      // 32      f32 = 128
#define OFF_BH    (OFF_BL  + 128)       // 16      f32 = 64  (bh padded)
#define OFF_JMP   (OFF_BH  + 64)        // 15      f32 -> 64
#define OFF_Y     (OFF_JMP + 64)        // 64*5    f32 = 1280
#define OFF_STG0  (OFF_Y   + 1280)      // 256     f32 = 1024 (192 u + 64 dt)
#define OFF_STG1  (OFF_STG0 + 1024)     // 256     f32 = 1024
#define OFF_TH    (OFF_STG1 + 1024)     // 64*16   f32 = 4096
#define SMEM_BYTES (OFF_TH + 4096)      // ~171.5 KB (< 320KB WGP LDS)

__device__ __forceinline__ float sigmoidf_(float x) {
  return 1.f / (1.f + __expf(-x));
}
__device__ __forceinline__ float tanhf_(float x) {
  return 2.f / (1.f + __expf(-2.f * x)) - 1.f;
}
__device__ __forceinline__ v8f splat8(float v) {
  v8f r;
#pragma unroll
  for (int i = 0; i < 8; ++i) r[i] = v;
  return r;
}

// 16-bit A matrix 16x32 (MxK) per CDNA5 ISA layout:
// lane L<16: row M=L, halves 0..7 -> K=0..7, halves 8..15 -> K=16..23
// lane L>=16: row M=L-16, K offset +8 (K=8..15 and 24..31)
__device__ __forceinline__ v16h load_A(const _Float16* base, int ld, int m0,
                                       int k0, int lane) {
  const int m = m0 + (lane & 15);
  const int koff = k0 + ((lane >> 4) << 3);
  const _Float16* p = base + m * ld + koff;
  v16h a;
#pragma unroll
  for (int i = 0; i < 8; ++i) a[i] = p[i];
#pragma unroll
  for (int i = 0; i < 8; ++i) a[8 + i] = p[16 + i];
  return a;
}

// B = W^T tile (32x16, KxN): lane L holds column N = j0 + L%16,
// K = k0 + (L/16)*8 + 0..7 and +16..23 (matches A). W is row-major (j, k).
__device__ __forceinline__ v16h load_B(const _Float16* w, int ldw, int j0,
                                       int k0, int lane) {
  const int j = j0 + (lane & 15);
  const int kk = k0 + ((lane >> 4) << 3);
  const _Float16* p = w + j * ldw + kk;
  v16h b;
#pragma unroll
  for (int i = 0; i < 8; ++i) b[i] = p[i];
#pragma unroll
  for (int i = 0; i < 8; ++i) b[8 + i] = p[16 + i];
  return b;
}

__device__ __forceinline__ v8f wmma(v16h a, v16h b, v8f c) {
  return __builtin_amdgcn_wmma_f32_16x16x32_f16(false, a, false, b, (short)0, c,
                                                false, false);
}

__device__ __forceinline__ void rhs5(const float* th, const float* y, float* d) {
  const float A = y[0], D = y[1], G = y[2], Jv = y[3], M = y[4];
  const float kf1 = th[0], kf2 = th[1], kf3 = th[2], kf4 = th[3];
  const float kr1 = th[4], kr2 = th[5], kr3 = th[6], kr4 = th[7];
  d[0] = -kf1 * A + kr1 * D + th[8];
  d[1] = kf1 * A - kr1 * D - kf2 * D + kr2 * G + th[9];
  d[2] = kf2 * D - kr2 * G - kf3 * G + kr3 * Jv + th[10];
  d[3] = kf3 * G - kr3 * Jv - kf4 * Jv + kr4 * M + th[11];
  d[4] = kf4 * Jv - kr4 * M + th[12];
}

// One async dword per lane: 192 u-elements + 64 dt-elements = 256 lanes.
// Each wave issues exactly one global_load_async_to_lds_b32 (ASYNCcnt += 1).
__device__ __forceinline__ void issue_stage(const float* u_seq,
                                            const float* dt_seq,
                                            const char* smem_base, float* stg,
                                            int b0, int k, int tid) {
  unsigned long long gaddr;
  if (tid < TM * U_IN) {
    const int m = tid / U_IN;
    const int c = tid - m * U_IN;
    gaddr = (unsigned long long)(uintptr_t)(
        u_seq + ((size_t)(b0 + m) * K_STEPS + k) * U_IN + c);
  } else {
    const int m = tid - TM * U_IN;
    gaddr = (unsigned long long)(uintptr_t)(dt_seq +
                                            (size_t)(b0 + m) * K_STEPS + k);
  }
  const unsigned lds_off =
      (unsigned)((const char*)(stg + tid) - smem_base);  // dynamic LDS base = 0
  asm volatile("global_load_async_to_lds_b32 %0, %1, off"
               :
               : "v"(lds_off), "v"(gaddr)
               : "memory");
}

__global__ __launch_bounds__(THREADS, 1) void gru_rk4_scan(
    const float* __restrict__ y0, const float* __restrict__ u_seq,
    const float* __restrict__ dt_seq, const float* __restrict__ Wl,
    const float* __restrict__ bl, const float* __restrict__ W_ih,
    const float* __restrict__ b_ih, const float* __restrict__ W_hh,
    const float* __restrict__ b_hh, const float* __restrict__ Wh,
    const float* __restrict__ bh, const float* __restrict__ jmp,
    float* __restrict__ y_out, float* __restrict__ th_out) {
  extern __shared__ char smem[];
  _Float16* whh_s = (_Float16*)(smem + OFF_WHH);
  _Float16* wih_s = (_Float16*)(smem + OFF_WIH);
  _Float16* whd_s = (_Float16*)(smem + OFF_WHD);
  _Float16* xs    = (_Float16*)(smem + OFF_X);
  _Float16* h0s   = (_Float16*)(smem + OFF_H0);
  _Float16* h1s   = (_Float16*)(smem + OFF_H1);
  float* wl_s  = (float*)(smem + OFF_WL);
  float* bih_s = (float*)(smem + OFF_BIH);
  float* bhh_s = (float*)(smem + OFF_BHH);
  float* bl_s  = (float*)(smem + OFF_BL);
  float* bh_s  = (float*)(smem + OFF_BH);
  float* jmp_s = (float*)(smem + OFF_JMP);
  float* y_s   = (float*)(smem + OFF_Y);
  float* stg0  = (float*)(smem + OFF_STG0);
  float* stg1  = (float*)(smem + OFF_STG1);
  float* th_s  = (float*)(smem + OFF_TH);

  const int tid = threadIdx.x;
  const int lane = tid & 31;
  const int wv = tid >> 5;
  const int b0 = blockIdx.x * TM;

  // ---- one-time preload: weights (f32 -> f16 into LDS), biases, state ----
  for (int i = tid; i < G3 * HID; i += THREADS) whh_s[i] = (_Float16)W_hh[i];
  for (int i = tid; i < G3 * LIFT; i += THREADS) wih_s[i] = (_Float16)W_ih[i];
  for (int i = tid; i < 16 * HID; i += THREADS)
    whd_s[i] = (i < 13 * HID) ? (_Float16)Wh[i] : (_Float16)0.f;
  for (int i = tid; i < LIFT * 8; i += THREADS) wl_s[i] = Wl[i];
  for (int i = tid; i < G3; i += THREADS) {
    bih_s[i] = b_ih[i];
    bhh_s[i] = b_hh[i];
  }
  for (int i = tid; i < LIFT; i += THREADS) bl_s[i] = bl[i];
  for (int i = tid; i < 16; i += THREADS) bh_s[i] = (i < 13) ? bh[i] : 0.f;
  for (int i = tid; i < U_IN * P_ST; i += THREADS) jmp_s[i] = jmp[i];
  for (int i = tid; i < TM * HID; i += THREADS) {
    h0s[i] = (_Float16)0.f;
    h1s[i] = (_Float16)0.f;
  }
  for (int i = tid; i < TM * P_ST; i += THREADS)
    y_s[i] = y0[(size_t)(b0 + i / P_ST) * P_ST + (i % P_ST)] + 0.01f;

  // prefetch step-0 u/dt into staging buffer 0 (async to LDS)
  issue_stage(u_seq, dt_seq, smem, stg0, b0, 0, tid);
  __syncthreads();

  // ---- loop-invariant hoists: wave wv owns N-column block n0 for all steps --
  const int n0 = wv * 16;
  const int jc = n0 + (lane & 15);
  const float biasr = bih_s[jc] + bhh_s[jc];
  const float biasz = bih_s[HID + jc] + bhh_s[HID + jc];
  const float biasx = bih_s[2 * HID + jc];
  const float biash = bhh_s[2 * HID + jc];

  // W_hh B-tiles for this wave's column block: resident in VGPRs for all 512
  // steps (12 x v16h = 96 VGPRs) -- removes the dominant per-step LDS traffic.
  v16h bhr[4], bhz[4], bhn[4];
#pragma unroll
  for (int kk = 0; kk < 4; ++kk) {
    bhr[kk] = load_B(whh_s, HID, n0, kk * 32, lane);
    bhz[kk] = load_B(whh_s, HID, HID + n0, kk * 32, lane);
    bhn[kk] = load_B(whh_s, HID, 2 * HID + n0, kk * 32, lane);
  }

  // hidden state carried in f32 registers (C layout), owned by this wave:
  // hreg[mt][e] <-> h[m0+e+8*(lane>=16)][n0+lane%16]
  v8f hreg[4];
#pragma unroll
  for (int p = 0; p < 4; ++p) hreg[p] = splat8(0.f);

  int cur = 0;
  for (int k = 0; k < K_STEPS; ++k) {
    float* u_s = (k & 1) ? stg1 : stg0;
    float* dt_s = u_s + TM * U_IN;

    // software pipeline: kick off step k+1 staging, then wait for step k batch
    if (k + 1 < K_STEPS) {
      issue_stage(u_seq, dt_seq, smem, (k & 1) ? stg0 : stg1, b0, k + 1, tid);
      asm volatile("s_wait_asynccnt 1" ::: "memory");
    } else {
      asm volatile("s_wait_asynccnt 0" ::: "memory");
    }
    __syncthreads();

    // ---- lift: x = silu([u, y] @ Wl^T + bl), 64x32 outputs ----
    for (int i = tid; i < TM * LIFT; i += THREADS) {
      const int m = i >> 5;
      const int c = i & 31;
      const float* wr = wl_s + c * 8;
      float acc = bl_s[c];
      acc += wr[0] * u_s[m * 3 + 0] + wr[1] * u_s[m * 3 + 1] +
             wr[2] * u_s[m * 3 + 2];
#pragma unroll
      for (int p = 0; p < P_ST; ++p) acc += wr[3 + p] * y_s[m * 5 + p];
      xs[m * LIFT + c] = (_Float16)(acc * sigmoidf_(acc));
    }
    __syncthreads();

    const _Float16* hsrc = (cur == 0) ? h0s : h1s;
    _Float16* hdst = (cur == 0) ? h1s : h0s;

    // ---- GRU: this wave computes its column block for all 4 M-tiles.
    // Per-k-slice ordering keeps only one A tile live at a time (low VGPR
    // pressure -> stay under 256 VGPRs, no s_set_vgpr_msb traffic).
#pragma unroll 1
    for (int mt = 0; mt < 4; ++mt) {
      const int m0 = mt * 16;

      v8f accr = splat8(biasr);
      v8f accz = splat8(biasz);
      v8f accx = splat8(biasx);  // gx_n (kept separate)
      v8f acch = splat8(biash);  // gh_n (kept separate)

      {
        const v16h ax = load_A(xs, LIFT, m0, 0, lane);
        accr = wmma(ax, load_B(wih_s, LIFT, n0, 0, lane), accr);
        accz = wmma(ax, load_B(wih_s, LIFT, HID + n0, 0, lane), accz);
        accx = wmma(ax, load_B(wih_s, LIFT, 2 * HID + n0, 0, lane), accx);
      }
#pragma unroll
      for (int kk = 0; kk < 4; ++kk) {
        const v16h ah = load_A(hsrc, HID, m0, kk * 32, lane);
        accr = wmma(ah, bhr[kk], accr);
        accz = wmma(ah, bhz[kk], accz);
        acch = wmma(ah, bhn[kk], acch);
      }

#pragma unroll
      for (int e = 0; e < 8; ++e) {
        const int m = m0 + e + ((lane >> 4) << 3);
        const int n = n0 + (lane & 15);
        const float r = sigmoidf_(accr[e]);
        const float z = sigmoidf_(accz[e]);
        const float nn = tanhf_(accx[e] + r * acch[e]);
        const float hnew = (1.f - z) * nn + z * hreg[mt][e];
        hreg[mt][e] = hnew;                     // f32 carry in registers
        hdst[m * HID + n] = (_Float16)hnew;     // f16 copy for WMMA A-side
      }
    }
    __syncthreads();

    // ---- head: raw = h_new @ Wh^T (13 padded to 16); waves 0..3 ----
    if (wv < 4) {
      const int m0 = wv * 16;
      const int j = lane & 15;
      v8f acc = splat8(bh_s[j]);
#pragma unroll
      for (int kk = 0; kk < 4; ++kk) {
        const v16h ah = load_A(hdst, HID, m0, kk * 32, lane);
        acc = wmma(ah, load_B(whd_s, HID, 0, kk * 32, lane), acc);
      }
#pragma unroll
      for (int e = 0; e < 8; ++e) {
        const int m = m0 + e + ((lane >> 4) << 3);
        const float s = sigmoidf_(acc[e]);
        const float th = (j < 8) ? (0.01f + 2.99f * s) : (0.3f * s);
        th_s[m * 16 + j] = th;
        if (j < 13)
          th_out[((size_t)(b0 + m) * K_STEPS + k) * 13 + j] = th;
      }
    }
    __syncthreads();

    // ---- RK4 integration: thread m (< 64) owns one batch row ----
    if (tid < TM) {
      const int m = tid;
      float th[13];
#pragma unroll
      for (int j = 0; j < 13; ++j) th[j] = th_s[m * 16 + j];
      float y[5];
#pragma unroll
      for (int i = 0; i < 5; ++i) y[i] = y_s[m * 5 + i];
      const float uu0 = u_s[m * 3 + 0], uu1 = u_s[m * 3 + 1],
                  uu2 = u_s[m * 3 + 2];
#pragma unroll
      for (int i = 0; i < 5; ++i)
        y[i] += uu0 * jmp_s[0 * 5 + i] + uu1 * jmp_s[1 * 5 + i] +
                uu2 * jmp_s[2 * 5 + i];
      const float h = dt_s[m] * (1.f / (float)NSUB);
#pragma unroll 1
      for (int s = 0; s < NSUB; ++s) {
        float k1[5], k2[5], k3[5], k4[5], yt[5];
        rhs5(th, y, k1);
#pragma unroll
        for (int i = 0; i < 5; ++i) yt[i] = y[i] + 0.5f * h * k1[i];
        rhs5(th, yt, k2);
#pragma unroll
        for (int i = 0; i < 5; ++i) yt[i] = y[i] + 0.5f * h * k2[i];
        rhs5(th, yt, k3);
#pragma unroll
        for (int i = 0; i < 5; ++i) yt[i] = y[i] + h * k3[i];
        rhs5(th, yt, k4);
#pragma unroll
        for (int i = 0; i < 5; ++i) {
          float v = y[i] + (h * (1.f / 6.f)) *
                               (k1[i] + 2.f * k2[i] + 2.f * k3[i] + k4[i]);
          y[i] = fmaxf(v, 0.f);
        }
      }
#pragma unroll
      for (int i = 0; i < 5; ++i) {
        y_s[m * 5 + i] = y[i];
        y_out[((size_t)(b0 + m) * K_STEPS + k) * 5 + i] = y[i];
      }
    }
    __syncthreads();
    cur ^= 1;
  }
}

extern "C" void kernel_launch(void* const* d_in, const int* in_sizes, int n_in,
                              void* d_out, int out_size, void* d_ws,
                              size_t ws_size, hipStream_t stream) {
  (void)in_sizes; (void)n_in; (void)out_size; (void)d_ws; (void)ws_size;
  const float* y0   = (const float*)d_in[0];
  const float* useq = (const float*)d_in[1];
  const float* dts  = (const float*)d_in[2];
  const float* Wl   = (const float*)d_in[3];
  const float* bl   = (const float*)d_in[4];
  const float* Wih  = (const float*)d_in[5];
  const float* bih  = (const float*)d_in[6];
  const float* Whh  = (const float*)d_in[7];
  const float* bhh  = (const float*)d_in[8];
  const float* Wh   = (const float*)d_in[9];
  const float* bh   = (const float*)d_in[10];
  const float* jmp  = (const float*)d_in[11];
  float* yout  = (float*)d_out;
  float* thout = yout + (size_t)B_TOT * K_STEPS * P_ST;

  hipLaunchKernelGGL(gru_rk4_scan, dim3(NBLK), dim3(THREADS), SMEM_BYTES,
                     stream, y0, useq, dts, Wl, bl, Wih, bih, Whh, bhh, Wh, bh,
                     jmp, yout, thout);
}
